// SequenceDenoiser_11355893531231
// MI455X (gfx1250) — matrix operand order repaired and checked by reading, not compiled, not run on hardware
//
#include <hip/hip_runtime.h>
#include <hip/hip_bf16.h>
#include <stdint.h>

typedef __attribute__((ext_vector_type(16))) __bf16 v16bf;
typedef __attribute__((ext_vector_type(8)))  __bf16 v8bf;
typedef __attribute__((ext_vector_type(4)))  __bf16 v4bf;
typedef __attribute__((ext_vector_type(8)))  float  v8f;

#define BB 4
#define NN 2048
#define KN 30
#define DD 128
#define USE_ASYNC_LDS 1

// ---------------------------------------------------------------- math utils
__device__ __forceinline__ float fast_tanh(float x) {
#if __has_builtin(__builtin_amdgcn_tanhf)
    return __builtin_amdgcn_tanhf(x);          // v_tanh_f32 (gfx1250 TRANS op)
#else
    float e = __expf(2.0f * x);                // branchless v_exp_f32 path
    return 1.0f - 2.0f / (e + 1.0f);
#endif
}

__device__ __forceinline__ float gelu_f(float x) {
    const float k0 = 0.7978845608028654f, k1 = 0.044715f;
    float u = k0 * (x + k1 * x * x * x);
    return 0.5f * x * (1.0f + fast_tanh(u));
}

__device__ __forceinline__ v8f wmma_bf16(v16bf a, v16bf b, v8f c) {
    return __builtin_amdgcn_wmma_f32_16x16x32_bf16(
        false, a, false, b, (short)0, c, false, false);
}

// A-fragment (16x32 bf16) from row-major bf16 (LDS tile); lane r = row M=r.
// ISA layout: lanes 0-15: K in {0..7,16..23}; lanes 16-31: K in {8..15,24..31}.
__device__ __forceinline__ v16bf load_a_frag_lds(const __bf16* row, int koff, int hi) {
    v8bf lo = *(const v8bf*)(row + koff + hi * 8);
    v8bf hh = *(const v8bf*)(row + koff + 16 + hi * 8);
    return __builtin_shufflevector(lo, hh, 0,1,2,3,4,5,6,7,8,9,10,11,12,13,14,15);
}

// B-fragment (32x16 bf16) from transposed bf16 weight WT[n][k]; two 16B loads, K order as A.
__device__ __forceinline__ v16bf load_b_frag(const __bf16* __restrict__ WT, int Kdim,
                                             int ctile, int r, int hi, int kbase) {
    const __bf16* p = WT + (size_t)(ctile * 16 + r) * Kdim + kbase + hi * 8;
    v8bf lo = *(const v8bf*)p;
    v8bf hh = *(const v8bf*)(p + 16);
    return __builtin_shufflevector(lo, hh, 0,1,2,3,4,5,6,7,8,9,10,11,12,13,14,15);
}

// 16B global -> LDS copy via CDNA5 async DMA (ASYNCcnt-tracked), see ISA 08 §4.
__device__ __forceinline__ void async_copy_b128(void* lds_dst, const void* gsrc) {
#if USE_ASYNC_LDS
    uint32_t lds_off = (uint32_t)(uintptr_t)lds_dst;   // generic LDS addr[31:0] = offset
    asm volatile("global_load_async_to_lds_b128 %0, %1, off"
                 :: "v"(lds_off), "v"(gsrc) : "memory");
#else
    *(v8bf*)lds_dst = *(const v8bf*)gsrc;
#endif
}
__device__ __forceinline__ void async_wait_all() {
#if USE_ASYNC_LDS
    asm volatile("s_wait_asynccnt 0" ::: "memory");
#endif
}

// ---------------------------------------------------------------- weight prep
__global__ __launch_bounds__(256) void cvt_tr_kernel(const float* __restrict__ W,
                                                     __bf16* __restrict__ WT, int K) {
    int idx = blockIdx.x * 256 + threadIdx.x;
    int total = K * DD;
    if (idx < total) {
        int n = idx / K, k = idx % K;
        WT[idx] = (__bf16)W[k * DD + n];
    }
}

__global__ __launch_bounds__(256) void cvt_bf16_vec4(const float* __restrict__ X,
                                                     __bf16* __restrict__ Y, int total4) {
    int i = blockIdx.x * 256 + threadIdx.x;
    if (i < total4) {
        float4 v = ((const float4*)X)[i];
        v4bf o;
        o[0] = (__bf16)v.x; o[1] = (__bf16)v.y; o[2] = (__bf16)v.z; o[3] = (__bf16)v.w;
        ((v4bf*)Y)[i] = o;
    }
}

// ------------------------------------------------- time features + adaLN mods
__global__ __launch_bounds__(256) void timemod_kernel(
    const float* __restrict__ t,
    const float* __restrict__ smW,  const float* __restrict__ smb,
    const float* __restrict__ a1W,  const float* __restrict__ a1b,
    const float* __restrict__ a2W,  const float* __restrict__ a2b,
    float* __restrict__ mod) {
    int g = blockIdx.x, b = blockIdx.y, tix = threadIdx.x;
    __shared__ float tf[DD];
    if (tix < DD) {
        int i = tix >> 1;
        float wn = powf(10000.0f, -((float)(2 * i)) / (float)DD);
        float ph = t[b] * wn;
        tf[tix] = (tix & 1) ? cosf(ph) : sinf(ph);
    }
    __syncthreads();
    const float* W; const float* bb;
    if (g == 0) { W = smW; bb = smb; }
    else {
        int l = (g - 1) >> 1;
        if (((g - 1) & 1) == 0) { W = a1W + (size_t)l * DD * 256; bb = a1b + l * 256; }
        else                    { W = a2W + (size_t)l * DD * 256; bb = a2b + l * 256; }
    }
    float acc = bb[tix];
    for (int k = 0; k < DD; ++k) acc += tf[k] * W[k * 256 + tix];
    mod[((size_t)g * BB + b) * 256 + tix] = acc;
}

// ------------------------------------------- per-node 2-layer MLP (+res+adaln)
template<int K1>
__global__ __launch_bounds__(256) void mlp_node_kernel(
    const float* __restrict__ X0, const float* __restrict__ X1,
    const float* __restrict__ Vres,
    const __bf16* __restrict__ W1T, const float* __restrict__ b1,
    const __bf16* __restrict__ W2T, const float* __restrict__ b2,
    const float* __restrict__ mod,
    float* __restrict__ Vout, __bf16* __restrict__ Voutb) {
    const int t = threadIdx.x, w = t >> 5, lane = t & 31, r = lane & 15, hi = lane >> 4;
    const size_t row0 = (size_t)blockIdx.x * 32;
    const int b = (int)(row0 >> 11);
    __shared__ union { __bf16 a[32][K1]; float o[32][DD]; } smu;
    __shared__ __bf16 hbuf[32][DD];

    // stage A tile (32 x K1) as bf16 into LDS, converting once per element
    constexpr int CPR = K1 / 8;
#pragma unroll
    for (int j = 0; j < (32 * CPR) / 256; ++j) {
        int cid = t + 256 * j;
        int lr = cid / CPR, co = (cid % CPR) * 8;
        const float* src = (co < DD) ? (X0 + (row0 + lr) * DD + co)
                                     : (X1 + (row0 + lr) * DD + (co - DD));
        v8bf d;
#pragma unroll
        for (int q = 0; q < 8; ++q) d[q] = (__bf16)src[q];
        *(v8bf*)&smu.a[lr][co] = d;
    }
    __syncthreads();

    v8f acc0 = {}, acc1 = {};
#pragma unroll
    for (int kk = 0; kk < K1 / 32; ++kk) {
        int kbase = kk * 32;
        v16bf bf = load_b_frag(W1T, K1, w, r, hi, kbase);
        v16bf a0 = load_a_frag_lds(&smu.a[r][0], kbase, hi);
        v16bf a1 = load_a_frag_lds(&smu.a[16 + r][0], kbase, hi);
        acc0 = wmma_bf16(a0, bf, acc0);
        acc1 = wmma_bf16(a1, bf, acc1);
    }
    const int col = w * 16 + r;
    {
        float bias = b1[col];
#pragma unroll
        for (int v = 0; v < 8; ++v) {
            int row = v + 8 * hi;
            hbuf[row][col]      = (__bf16)gelu_f(acc0[v] + bias);
            hbuf[row + 16][col] = (__bf16)gelu_f(acc1[v] + bias);
        }
    }
    __syncthreads();   // all A reads done before smu.o overlay is written
    v8f o0 = {}, o1 = {};
#pragma unroll
    for (int kk = 0; kk < 4; ++kk) {
        int kbase = kk * 32;
        v16bf bf = load_b_frag(W2T, DD, w, r, hi, kbase);
        v16bf a0 = load_a_frag_lds(&hbuf[r][0], kbase, hi);
        v16bf a1 = load_a_frag_lds(&hbuf[16 + r][0], kbase, hi);
        o0 = wmma_bf16(a0, bf, o0);
        o1 = wmma_bf16(a1, bf, o1);
    }
    {
        float bias = b2[col];
#pragma unroll
        for (int v = 0; v < 8; ++v) {
            int row = v + 8 * hi;
            smu.o[row][col]      = o0[v] + bias;
            smu.o[row + 16][col] = o1[v] + bias;
        }
    }
    __syncthreads();
    if (t < 32) {
        const float* vr = Vres + (row0 + t) * DD;
        const float* oo = smu.o[t];
        float mean = 0.f;
        for (int c = 0; c < DD; ++c) mean += oo[c] + vr[c];
        mean *= (1.0f / DD);
        float var = 0.f;
        for (int c = 0; c < DD; ++c) { float d = oo[c] + vr[c] - mean; var += d * d; }
        var *= (1.0f / DD);
        float rs = rsqrtf(var + 1e-5f);
        float*  out  = Vout  + (row0 + t) * DD;
        __bf16* outb = Voutb + (row0 + t) * DD;
        for (int c = 0; c < DD; ++c) {
            float g  = mod[b * 256 + c];
            float bt = mod[b * 256 + DD + c];
            float y  = (oo[c] + vr[c] - mean) * rs * (1.0f + g) + bt;
            out[c] = y;
            outb[c] = (__bf16)y;
        }
    }
}

// --------------------------------------------- per-edge message MLP + gather
// One workgroup = 2 nodes (64 edge rows); 8 waves, wave = 1 col-tile x 4 M-tiles.
__global__ __launch_bounds__(256) void msg_kernel(
    const float* __restrict__ Vin, const __bf16* __restrict__ Vinb,
    const __bf16* __restrict__ Eb, const int* __restrict__ Kidx,
    const __bf16* __restrict__ W1T, const float* __restrict__ b1,
    const __bf16* __restrict__ W2T, const float* __restrict__ b2,
    const float* __restrict__ mod,
    float* __restrict__ Vout, __bf16* __restrict__ Voutb) {
    const int b = blockIdx.y;
    const int node0 = blockIdx.x * 2;
    const int t = threadIdx.x, w = t >> 5, lane = t & 31, r = lane & 15, hi = lane >> 4;
    __shared__ union { __bf16 a[64][3 * DD]; float o[64][DD]; } smu;
    __shared__ __bf16 hbuf[64][DD];
    __shared__ float red[256];
    __shared__ int ksh[64];

    if (t < 64) {
        int nl = t >> 5, e = t & 31; if (e >= KN) e = KN - 1;
        ksh[t] = Kidx[((size_t)b * NN + node0 + nl) * KN + e];
    }
    __syncthreads();

    // Vi block (cols 0..127): 2 distinct rows, replicate across 32 edge rows each
    if (t < 32) {
        int nl = t >> 4, ch = t & 15;
        v8bf v = *(const v8bf*)(Vinb + ((size_t)b * NN + node0 + nl) * DD + ch * 8);
        for (int q = 0; q < 32; ++q)
            *(v8bf*)&smu.a[nl * 32 + q][ch * 8] = v;
    }
    // E block (cols 128..255): async DMA 16B chunks straight into LDS
#pragma unroll
    for (int j = 0; j < 4; ++j) {
        int cid = t + 256 * j;                 // 1024 chunks
        int lr = cid >> 4, ch = cid & 15;
        int nl = lr >> 5, e = lr & 31; if (e >= KN) e = KN - 1;
        const __bf16* src = Eb + (((size_t)b * NN + node0 + nl) * KN + e) * DD + ch * 8;
        async_copy_b128(&smu.a[lr][DD + ch * 8], src);
    }
    // Vj gather block (cols 256..383)
#pragma unroll
    for (int j = 0; j < 4; ++j) {
        int cid = t + 256 * j;
        int lr = cid >> 4, ch = cid & 15;
        const __bf16* src = Vinb + ((size_t)b * NN + ksh[lr]) * DD + ch * 8;
        async_copy_b128(&smu.a[lr][2 * DD + ch * 8], src);
    }
    async_wait_all();
    __syncthreads();

    v8f acc[4] = {};
#pragma unroll
    for (int kk = 0; kk < 12; ++kk) {          // K = 3*D = 384
        int kbase = kk * 32;
        v16bf bf = load_b_frag(W1T, 3 * DD, w, r, hi, kbase);
#pragma unroll
        for (int mt = 0; mt < 4; ++mt) {
            v16bf a = load_a_frag_lds(&smu.a[mt * 16 + r][0], kbase, hi);
            acc[mt] = wmma_bf16(a, bf, acc[mt]);
        }
    }
    const int col = w * 16 + r;
    {
        float bias = b1[col];
#pragma unroll
        for (int mt = 0; mt < 4; ++mt)
#pragma unroll
            for (int v = 0; v < 8; ++v)
                hbuf[mt * 16 + v + 8 * hi][col] = (__bf16)gelu_f(acc[mt][v] + bias);
    }
    __syncthreads();   // A-tile reads complete before smu.o overlay
    v8f o[4] = {};
#pragma unroll
    for (int kk = 0; kk < 4; ++kk) {
        int kbase = kk * 32;
        v16bf bf = load_b_frag(W2T, DD, w, r, hi, kbase);
#pragma unroll
        for (int mt = 0; mt < 4; ++mt) {
            v16bf a = load_a_frag_lds(&hbuf[mt * 16 + r][0], kbase, hi);
            o[mt] = wmma_bf16(a, bf, o[mt]);
        }
    }
    {
        float bias = b2[col];
#pragma unroll
        for (int mt = 0; mt < 4; ++mt)
#pragma unroll
            for (int v = 0; v < 8; ++v)
                smu.o[mt * 16 + v + 8 * hi][col] = o[mt][v] + bias;
    }
    __syncthreads();

    // aggregate 30 real edges per node, residual, LN + modulation (2 nodes parallel)
    const int nl = t >> 7, c = t & 127;
    const size_t gnode = (size_t)b * NN + node0 + nl;
    float s = 0.f;
#pragma unroll
    for (int e = 0; e < KN; ++e) s += smu.o[nl * 32 + e][c];
    float y = Vin[gnode * DD + c] + s * (1.0f / KN);
    red[t] = y;
    __syncthreads();
    for (int sft = 64; sft > 0; sft >>= 1) {
        if (c < sft) red[t] += red[t + sft];
        __syncthreads();
    }
    float mean = red[nl * 128] * (1.0f / DD);
    __syncthreads();
    { float d = y - mean; red[t] = d * d; }
    __syncthreads();
    for (int sft = 64; sft > 0; sft >>= 1) {
        if (c < sft) red[t] += red[t + sft];
        __syncthreads();
    }
    float var = red[nl * 128] * (1.0f / DD);
    float rs = rsqrtf(var + 1e-5f);
    float g  = mod[b * 256 + c];
    float bt = mod[b * 256 + DD + c];
    float outv = (y - mean) * rs * (1.0f + g) + bt;
    Vout[gnode * DD + c]  = outv;
    Voutb[gnode * DD + c] = (__bf16)outv;
}

// -------------------------------------------------------------- final proj
__global__ __launch_bounds__(256) void proj_kernel(const __bf16* __restrict__ Xb,
                                                   const __bf16* __restrict__ WT,
                                                   float* __restrict__ Out) {
    const int t = threadIdx.x, w = t >> 5, lane = t & 31, r = lane & 15, hi = lane >> 4;
    const size_t row0 = (size_t)blockIdx.x * 32;
    __shared__ __bf16 abuf[32][DD];
#pragma unroll
    for (int j = 0; j < 2; ++j) {
        int cid = t + 256 * j;
        int lr = cid >> 4, ch = cid & 15;
        *(v8bf*)&abuf[lr][ch * 8] = *(const v8bf*)(Xb + (row0 + lr) * DD + ch * 8);
    }
    __syncthreads();
    v8f acc0 = {}, acc1 = {};
#pragma unroll
    for (int kk = 0; kk < 4; ++kk) {
        int kbase = kk * 32;
        v16bf bf = load_b_frag(WT, DD, w, r, hi, kbase);
        v16bf a0 = load_a_frag_lds(&abuf[r][0], kbase, hi);
        v16bf a1 = load_a_frag_lds(&abuf[16 + r][0], kbase, hi);
        acc0 = wmma_bf16(a0, bf, acc0);
        acc1 = wmma_bf16(a1, bf, acc1);
    }
    const int col = w * 16 + r;
#pragma unroll
    for (int v = 0; v < 8; ++v) {
        Out[(row0 + v + 8 * hi) * DD + col]      = acc0[v];
        Out[(row0 + 16 + v + 8 * hi) * DD + col] = acc1[v];
    }
}

extern "C" void kernel_launch(void* const* d_in, const int* in_sizes, int n_in,
                              void* d_out, int out_size, void* d_ws, size_t ws_size,
                              hipStream_t stream) {
    const float* V       = (const float*)d_in[0];
    const float* E       = (const float*)d_in[1];
    const int*   Kidx    = (const int*)d_in[2];
    const float* Z       = (const float*)d_in[3];
    const float* t       = (const float*)d_in[4];
    const float* sm_W1   = (const float*)d_in[6];
    const float* sm_b1   = (const float*)d_in[7];
    const float* sm_W2   = (const float*)d_in[8];
    const float* sm_b2   = (const float*)d_in[9];
    const float* sm_adaW = (const float*)d_in[10];
    const float* sm_adab = (const float*)d_in[11];
    const float* msg_W1  = (const float*)d_in[12];
    const float* msg_b1  = (const float*)d_in[13];
    const float* msg_W2  = (const float*)d_in[14];
    const float* msg_b2  = (const float*)d_in[15];
    const float* ada1_W  = (const float*)d_in[16];
    const float* ada1_b  = (const float*)d_in[17];
    const float* ffn_W1  = (const float*)d_in[18];
    const float* ffn_b1  = (const float*)d_in[19];
    const float* ffn_W2  = (const float*)d_in[20];
    const float* ffn_b2  = (const float*)d_in[21];
    const float* ada2_W  = (const float*)d_in[22];
    const float* ada2_b  = (const float*)d_in[23];
    const float* proj_W  = (const float*)d_in[24];

    char* ws = (char*)d_ws;
    size_t o_smW1T  = 0;                      // 128*256 bf16
    size_t o_smW2T  = o_smW1T + 65536;
    size_t o_msgW1T = o_smW2T + 32768;        // 3*128*384 bf16
    size_t o_msgW2T = o_msgW1T + 294912;
    size_t o_ffnW1T = o_msgW2T + 98304;
    size_t o_ffnW2T = o_ffnW1T + 98304;
    size_t o_projT  = o_ffnW2T + 98304;
    size_t o_mod    = o_projT + 32768;        // 7*B*256 f32
    size_t o_Vcur   = o_mod + 28672;
    size_t o_Vtmp   = o_Vcur + (size_t)BB * NN * DD * 4;
    size_t o_Vcurb  = o_Vtmp + (size_t)BB * NN * DD * 4;
    size_t o_Vtmpb  = o_Vcurb + (size_t)BB * NN * DD * 2;
    size_t o_Eb     = o_Vtmpb + (size_t)BB * NN * DD * 2;   // B*N*KN*D bf16

    __bf16* smW1T  = (__bf16*)(ws + o_smW1T);
    __bf16* smW2T  = (__bf16*)(ws + o_smW2T);
    __bf16* msgW1T = (__bf16*)(ws + o_msgW1T);
    __bf16* msgW2T = (__bf16*)(ws + o_msgW2T);
    __bf16* ffnW1T = (__bf16*)(ws + o_ffnW1T);
    __bf16* ffnW2T = (__bf16*)(ws + o_ffnW2T);
    __bf16* projT  = (__bf16*)(ws + o_projT);
    float*  mod    = (float*)(ws + o_mod);
    float*  Vcur   = (float*)(ws + o_Vcur);
    float*  Vtmp   = (float*)(ws + o_Vtmp);
    __bf16* Vcurb  = (__bf16*)(ws + o_Vcurb);
    __bf16* Vtmpb  = (__bf16*)(ws + o_Vtmpb);
    __bf16* Eb     = (__bf16*)(ws + o_Eb);

    auto cvt = [&](const float* W, __bf16* WT, int K) {
        int total = K * DD;
        cvt_tr_kernel<<<(total + 255) / 256, 256, 0, stream>>>(W, WT, K);
    };
    cvt(sm_W1, smW1T, 256);
    cvt(sm_W2, smW2T, 128);
    for (int l = 0; l < 3; ++l) {
        cvt(msg_W1 + (size_t)l * 384 * DD, msgW1T + (size_t)l * DD * 384, 384);
        cvt(msg_W2 + (size_t)l * DD * DD,  msgW2T + (size_t)l * DD * DD,  128);
        cvt(ffn_W1 + (size_t)l * DD * DD,  ffnW1T + (size_t)l * DD * DD,  128);
        cvt(ffn_W2 + (size_t)l * DD * DD,  ffnW2T + (size_t)l * DD * DD,  128);
    }
    cvt(proj_W, projT, 128);

    {   // E -> bf16 once (read 3x later through L2)
        int total4 = BB * NN * KN * DD / 4;
        cvt_bf16_vec4<<<(total4 + 255) / 256, 256, 0, stream>>>(E, Eb, total4);
    }

    timemod_kernel<<<dim3(7, BB), 256, 0, stream>>>(t, sm_adaW, sm_adab,
                                                    ada1_W, ada1_b, ada2_W, ada2_b, mod);

    const int nrowblk = BB * NN / 32;  // 256

    mlp_node_kernel<256><<<nrowblk, 256, 0, stream>>>(V, Z, V, smW1T, sm_b1,
                                                      smW2T, sm_b2, mod, Vcur, Vcurb);

    float*  Va  = Vcur;  __bf16* Vab = Vcurb;
    float*  Vb  = Vtmp;  __bf16* Vbb = Vtmpb;
    for (int l = 0; l < 3; ++l) {
        msg_kernel<<<dim3(NN / 2, BB), 256, 0, stream>>>(
            Va, Vab, Eb, Kidx,
            msgW1T + (size_t)l * DD * 384, msg_b1 + l * DD,
            msgW2T + (size_t)l * DD * DD,  msg_b2 + l * DD,
            mod + (size_t)(1 + 2 * l) * BB * 256, Vb, Vbb);
        mlp_node_kernel<128><<<nrowblk, 256, 0, stream>>>(
            Vb, nullptr, Vb,
            ffnW1T + (size_t)l * DD * DD, ffn_b1 + l * DD,
            ffnW2T + (size_t)l * DD * DD, ffn_b2 + l * DD,
            mod + (size_t)(2 + 2 * l) * BB * 256, Va, Vab);
    }

    proj_kernel<<<nrowblk, 256, 0, stream>>>(Vab, projT, (float*)d_out);
}
